// Net_41317585388042
// MI455X (gfx1250) — matrix operand order repaired and checked by reading, not compiled, not run on hardware
//
#include <hip/hip_runtime.h>

typedef __attribute__((ext_vector_type(16))) __bf16 v16bf;
typedef __attribute__((ext_vector_type(8)))  float  v8f;
typedef __attribute__((ext_vector_type(4)))  unsigned int v4u;

#define KTOT 2304   // 9 taps * 256 input channels
#define KSTEPS 72   // 2304 / 32

__device__ __forceinline__ unsigned short f2bf(float f) {
    unsigned int u = __float_as_uint(f);
    unsigned int r = u + 0x7FFFu + ((u >> 16) & 1u);   // round-to-nearest-even
    return (unsigned short)(r >> 16);
}

// 32-bit LDS byte offset of a shared-memory pointer (flat LDS aperture keeps
// the offset in the low 32 bits)
__device__ __forceinline__ unsigned lds_off(const void* p) {
    return (unsigned)(size_t)p;
}

// ---------------------------------------------------------------------------
// Weight prep: w[Kout][256][3][3] (f32) -> bf16 matrix [rowOff+Kout][2304],
// K ordered as k = tap*256 + cin  (tap = dy*3+dx)
// ---------------------------------------------------------------------------
__global__ void k_prep_w(const float* __restrict__ w, unsigned short* __restrict__ o,
                         int Kout, int rowOff) {
    int i = blockIdx.x * blockDim.x + threadIdx.x;
    int total = Kout * KTOT;
    if (i >= total) return;
    int ko = i / KTOT;
    int k  = i - ko * KTOT;
    int r  = k >> 8;          // tap 0..8
    int c  = k & 255;         // cin
    o[(size_t)(rowOff + ko) * KTOT + k] = f2bf(w[(size_t)(ko * 256 + c) * 9 + r]);
}

__global__ void k_zero_u16(unsigned short* p, int n) {
    int i = blockIdx.x * blockDim.x + threadIdx.x;
    if (i < n) p[i] = 0;
}

__global__ void k_prep_bn(const float* g, const float* b, const float* m,
                          const float* v, float* inv, float* beta) {
    int c = blockIdx.x * blockDim.x + threadIdx.x;
    if (c >= 256) return;
    float iv = g[c] * rsqrtf(v[c] + 1e-5f);
    inv[c]  = iv;
    beta[c] = b[c] - m[c] * iv;
}

// out = a + upsample2x(b), all NCHW, C2 = B*C
__global__ void k_upadd(const float* __restrict__ a, const float* __restrict__ b,
                        float* __restrict__ out, int C2, int H, int W) {
    size_t i = (size_t)blockIdx.x * blockDim.x + threadIdx.x;
    size_t total = (size_t)C2 * H * W;
    if (i >= total) return;
    int x  = (int)(i % W);
    int y  = (int)((i / W) % H);
    size_t nc = i / ((size_t)H * W);
    out[i] = a[i] + b[(nc * (H >> 1) + (y >> 1)) * (W >> 1) + (x >> 1)];
}

// f32 NCHW -> bf16 channel-last [B][HW][256]
__global__ void k_tobf16t(const float* __restrict__ in, unsigned short* __restrict__ o,
                          int HW) {
    size_t i = (size_t)blockIdx.x * blockDim.x + threadIdx.x;
    size_t per = (size_t)256 * HW;
    if (i >= 2 * per) return;
    size_t img = i / per;
    size_t rem = i - img * per;
    int c = (int)(rem / HW);
    int p = (int)(rem - (size_t)c * HW);
    o[(img * HW + p) * 256 + c] = f2bf(in[i]);
}

// ---------------------------------------------------------------------------
// Implicit-GEMM 3x3 conv via v_wmma_f32_16x16x32_bf16.
// Activations are bf16 channel-last [B][HW][256]; weights bf16 [Kout][2304].
// All A/B fragment data is staged with GLOBAL_LOAD_ASYNC_TO_LDS_B128 (each
// 16B chunk is 8 consecutive K = contiguous in memory for both operands);
// ASYNCcnt + one barrier fences the double-buffered LDS per K-step.
// OOB halo chunks are redirected to a zeroed 16B scratch block.
// ---------------------------------------------------------------------------
template <int MTILE>
__global__ void conv3x3_wmma(const unsigned short* __restrict__ inT,  // [B][HW][256] bf16
                             const unsigned short* __restrict__ Wbf,  // [Kout][2304] bf16
                             const unsigned short* __restrict__ zbuf, // 16B of zeros
                             const float* __restrict__ bninv,
                             const float* __restrict__ bnbeta,
                             float* __restrict__ outF,                // optional f32 NCHW
                             unsigned short* __restrict__ outBf,      // optional bf16 chan-last
                             int H, int W, int lgW, int HW, int Kout, int act) {
    constexpr int BLK = (MTILE / 16) * 64;
    __shared__ __align__(32) unsigned short Alds[2][MTILE * 32];
    __shared__ __align__(32) unsigned short Blds[2][32 * 32];

    const int tid  = threadIdx.x;
    const int lane = tid & 31;
    const int wv   = tid >> 5;
    const int mi   = wv >> 1;
    const int ni   = wv & 1;
    const int p0   = blockIdx.x * 32;
    const int m0   = blockIdx.y * MTILE;
    const int img  = blockIdx.z;
    const unsigned short* inImg = inT + (size_t)img * HW * 256;

    auto stage = [&](int step, int buf) {
        const int k0 = step * 32;
        const int r  = k0 >> 8;        // tap
        const int dy = r / 3 - 1;
        const int dx = r % 3 - 1;
        const int c0 = k0 & 255;
        // ---- A (weights): MTILE*4 chunks of 16B, each = 8 consecutive K ----
        for (int ch = tid; ch < MTILE * 4; ch += BLK) {
            int h  = ch & 1;
            int q  = ch >> 1;
            int ll = q & 31;
            int m  = (q >> 5) * 16 + (ll & 15);
            int ks = ((ll >> 4) << 3) + (h << 4);     // grp*8 + h*16
            const unsigned short* g = Wbf + (size_t)(m0 + m) * KTOT + k0 + ks;
            unsigned lo = lds_off(&Alds[buf][ch << 3]);
            asm volatile("global_load_async_to_lds_b128 %0, %1, off"
                         :: "v"(lo), "v"(g) : "memory");
        }
        // ---- B (im2col): 128 chunks of 16B, each = 8 consecutive channels --
        for (int ch = tid; ch < 128; ch += BLK) {
            int h  = ch & 1;
            int q  = ch >> 1;
            int ll = q & 31;
            int n  = (q >> 5) * 16 + (ll & 15);
            int kb = ((ll >> 4) << 3) + (h << 4);     // grp*8 + h*16
            int p  = p0 + n;
            int py = (p >> lgW) + dy;
            int px = (p & (W - 1)) + dx;
            const unsigned short* g =
                (py >= 0 && py < H && px >= 0 && px < W)
                    ? inImg + ((size_t)((py << lgW) + px) * 256 + c0 + kb)
                    : zbuf;
            unsigned lo = lds_off(&Blds[buf][ch << 3]);
            asm volatile("global_load_async_to_lds_b128 %0, %1, off"
                         :: "v"(lo), "v"(g) : "memory");
        }
    };

    v8f acc = {0.f, 0.f, 0.f, 0.f, 0.f, 0.f, 0.f, 0.f};

    stage(0, 0);
#pragma unroll 1
    for (int step = 0; step < KSTEPS; ++step) {
        const int cur = step & 1;
        asm volatile("s_wait_asynccnt 0" ::: "memory");  // my staged chunks landed
        __syncthreads();                                 // everyone's landed; prev reads done
        if (step + 1 < KSTEPS) stage(step + 1, cur ^ 1);
        v16bf a = *(const v16bf*)&Alds[cur][((mi * 32 + lane) << 4)];
        v16bf b = *(const v16bf*)&Blds[cur][((ni * 32 + lane) << 4)];
        acc = __builtin_amdgcn_wmma_f32_16x16x32_bf16(
            false, a, false, b, (short)0, acc, false, false);
    }

    // ---- epilogue: fold BN + LeakyReLU, scatter per C/D fragment layout -----
    const int n     = p0 + ni * 16 + (lane & 15);
    const int mbase = m0 + mi * 16 + ((lane >> 4) << 3);
    v4u pk = {0u, 0u, 0u, 0u};
#pragma unroll
    for (int rr = 0; rr < 8; ++rr) {
        int m = mbase + rr;
        float v = acc[rr];
        if (act) {
            v = v * bninv[m] + bnbeta[m];
            v = (v >= 0.f) ? v : 0.01f * v;
        }
        if (outF) outF[((size_t)img * Kout + m) * HW + n] = v;
        pk[rr >> 1] |= (unsigned)f2bf(v) << ((rr & 1) << 4);
    }
    if (outBf)   // lane's 8 consecutive output channels: one aligned 16B store
        *(v4u*)&outBf[((size_t)img * HW + n) * 256 + mbase] = pk;
}

// ---------------------------------------------------------------------------
// Decode: sigmoid scores, box decode + clip, validity mask
// ---------------------------------------------------------------------------
__global__ void k_decode(const float* __restrict__ rpn,     // [B][16][HW]
                         const float* __restrict__ anchors, // [A][4]
                         float* __restrict__ scoreBuf,      // [B][A]
                         float* __restrict__ boxBuf,        // [B][A][4]
                         int HW, float stride) {
    int A = HW * 3;
    int i = blockIdx.x * blockDim.x + threadIdx.x;
    int img = blockIdx.y;
    if (i >= A) return;
    int p = i / 3, a = i - p * 3;
    const float* R = rpn + (size_t)img * 16 * HW;
    float sc = 1.f / (1.f + expf(-R[a * HW + p]));
    float t0 = R[(3 + 4 * a + 0) * HW + p];
    float t1 = R[(3 + 4 * a + 1) * HW + p];
    float t2 = R[(3 + 4 * a + 2) * HW + p];
    float t3 = R[(3 + 4 * a + 3) * HW + p];
    const float* an = anchors + (size_t)i * 4;
    float aw = an[2] - an[0], ah = an[3] - an[1];
    float x1 = an[0] + t0 * aw, y1 = an[1] + t1 * ah;
    float x2 = x1 + aw * expf(t2), y2 = y1 + ah * expf(t3);
    x1 = fminf(fmaxf(x1, 0.f), 512.f);
    y1 = fminf(fmaxf(y1, 0.f), 512.f);
    x2 = fminf(fmaxf(x2, 0.f), 512.f);
    y2 = fminf(fmaxf(y2, 0.f), 512.f);
    bool valid = (sc > 0.5f) && ((x2 - x1) >= 2.f * stride) && ((y2 - y1) >= 2.f * stride);
    scoreBuf[(size_t)img * A + i] = valid ? sc : -1.0f;
    float* bo = boxBuf + ((size_t)img * A + i) * 4;
    bo[0] = x1; bo[1] = y1; bo[2] = x2; bo[3] = y2;
}

// ---------------------------------------------------------------------------
// Top-256 (stable), NMS(+1 IoU), ordered selection of 32 dets.
// One block of 256 threads per (image); launched per level.
// ---------------------------------------------------------------------------
__global__ void k_topk_nms(float* __restrict__ scoreBuf, const float* __restrict__ boxBuf,
                           int A, int lvl, int sshift,
                           float* __restrict__ dets, float* __restrict__ maskOut,
                           int* __restrict__ coords, int* __restrict__ mflags) {
    __shared__ float sval[256]; __shared__ int sidx[256];
    __shared__ float tv[256];   __shared__ int tix[256];
    __shared__ float bx[256][4];
    __shared__ int   keep[256];
    const int tid = threadIdx.x;
    const int img = blockIdx.x;
    float* S = scoreBuf + (size_t)img * A;
    const float* BB = boxBuf + (size_t)img * A * 4;

    // iterative argmax (lowest index wins ties -> matches stable top_k)
    for (int t = 0; t < 256; ++t) {
        float bv = -1e30f; int bi = 0x7fffffff;
        for (int i = tid; i < A; i += 256) {
            float v = S[i];
            if (v > bv) { bv = v; bi = i; }
        }
        sval[tid] = bv; sidx[tid] = bi;
        __syncthreads();
        for (int off = 128; off > 0; off >>= 1) {
            if (tid < off) {
                if (sval[tid + off] > sval[tid] ||
                    (sval[tid + off] == sval[tid] && sidx[tid + off] < sidx[tid])) {
                    sval[tid] = sval[tid + off]; sidx[tid] = sidx[tid + off];
                }
            }
            __syncthreads();
        }
        if (tid == 0) { tv[t] = sval[0]; tix[t] = sidx[0]; S[sidx[0]] = -1e30f; }
        __syncthreads();
    }

    int gi = tix[tid];
    bx[tid][0] = BB[gi * 4 + 0]; bx[tid][1] = BB[gi * 4 + 1];
    bx[tid][2] = BB[gi * 4 + 2]; bx[tid][3] = BB[gi * 4 + 3];
    keep[tid] = (tv[tid] > 0.5f) ? 1 : 0;
    __syncthreads();

    float ax1 = bx[tid][0], ay1 = bx[tid][1], ax2 = bx[tid][2], ay2 = bx[tid][3];
    float areaJ = (ax2 - ax1 + 1.f) * (ay2 - ay1 + 1.f);
    for (int i = 0; i < 256; ++i) {
        if (keep[i] && tid > i && keep[tid]) {
            float xx1 = fmaxf(bx[i][0], ax1), yy1 = fmaxf(bx[i][1], ay1);
            float xx2 = fminf(bx[i][2], ax2), yy2 = fminf(bx[i][3], ay2);
            float inter = fmaxf(0.f, xx2 - xx1 + 1.f) * fmaxf(0.f, yy2 - yy1 + 1.f);
            float areaI = (bx[i][2] - bx[i][0] + 1.f) * (bx[i][3] - bx[i][1] + 1.f);
            if (inter / (areaI + areaJ - inter) > 0.3f) keep[tid] = 0;
        }
        __syncthreads();
    }

    if (tid == 0) {
        int order[32]; int cnt = 0;
        for (int j = 0; j < 256 && cnt < 32; ++j) if (keep[j])  order[cnt++] = j;
        for (int j = 0; j < 256 && cnt < 32; ++j) if (!keep[j]) order[cnt++] = j;
        for (int s = 0; s < 32; ++s) {
            int j = order[s];
            int m = keep[j];
            int slot = (img * 3 + lvl) * 32 + s;
            float* dd = dets + (size_t)slot * 5;
            if (m) {
                dd[0] = bx[j][0]; dd[1] = bx[j][1]; dd[2] = bx[j][2];
                dd[3] = bx[j][3]; dd[4] = tv[j];
            } else {
                dd[0] = dd[1] = dd[2] = dd[3] = dd[4] = 0.f;
            }
            maskOut[slot] = m ? 1.f : 0.f;
            int* cc = coords + slot * 4;
            if (m) {
                cc[0] = ((int)bx[j][0]) >> sshift; cc[1] = ((int)bx[j][1]) >> sshift;
                cc[2] = ((int)bx[j][2]) >> sshift; cc[3] = ((int)bx[j][3]) >> sshift;
            } else {
                cc[0] = 0; cc[1] = 0; cc[2] = 2; cc[3] = 2;
            }
            mflags[slot] = m;
        }
    }
}

// ---------------------------------------------------------------------------
// Adaptive 7x7 max-pool over ROI (empty bins -> -FLT_MAX, masked ROIs -> 0)
// grid (32 slots, 2 imgs), 256 threads = channels; launched per level.
// ---------------------------------------------------------------------------
__global__ void k_roipool(const float* __restrict__ feat,  // [B][256][H*W]
                          const int* __restrict__ coords, const int* __restrict__ mflags,
                          float* __restrict__ rois, int H, int W, int lvl) {
    const int s = blockIdx.x, img = blockIdx.y, c = threadIdx.x;
    const int slot = (img * 3 + lvl) * 32 + s;
    float* out = rois + ((size_t)slot * 256 + c) * 49;
    if (!mflags[slot]) {
        for (int b = 0; b < 49; ++b) out[b] = 0.f;
        return;
    }
    const int* cc = coords + slot * 4;
    const int lx = cc[0], ly = cc[1];
    const int Lx = cc[2] - lx, Ly = cc[3] - ly;
    const float* F = feat + ((size_t)img * 256 + c) * H * W;
    for (int i = 0; i < 7; ++i) {
        int ys = max(ly + (i * Ly) / 7, 0);
        int ye = min(ly + ((i + 1) * Ly + 6) / 7, H);
        for (int j = 0; j < 7; ++j) {
            int xs = max(lx + (j * Lx) / 7, 0);
            int xe = min(lx + ((j + 1) * Lx + 6) / 7, W);
            float mv = -3.4028234663852886e+38f;
            for (int y = ys; y < ye; ++y)
                for (int x = xs; x < xe; ++x)
                    mv = fmaxf(mv, F[y * W + x]);
            out[i * 7 + j] = mv;
        }
    }
}

// ---------------------------------------------------------------------------
extern "C" void kernel_launch(void* const* d_in, const int* in_sizes, int n_in,
                              void* d_out, int out_size, void* d_ws, size_t ws_size,
                              hipStream_t stream) {
    const float* layer2 = (const float*)d_in[0];
    const float* layer3 = (const float*)d_in[1];
    const float* layer4 = (const float*)d_in[2];
    const float* anch[3] = {(const float*)d_in[3], (const float*)d_in[4], (const float*)d_in[5]};
    const float* hw1 = (const float*)d_in[6];
    const float* hg1 = (const float*)d_in[7];  const float* hb1 = (const float*)d_in[8];
    const float* hm1 = (const float*)d_in[9];  const float* hv1 = (const float*)d_in[10];
    const float* hw2 = (const float*)d_in[11];
    const float* hg2 = (const float*)d_in[12]; const float* hb2 = (const float*)d_in[13];
    const float* hm2 = (const float*)d_in[14]; const float* hv2 = (const float*)d_in[15];
    const float* clsw = (const float*)d_in[16];
    const float* locw = (const float*)d_in[17];

    // d_out layout: dets(2*3*32*5) | mask(2*3*32) | rois(2*3*32*256*49)
    float* dets = (float*)d_out;
    float* maskO = dets + 960;
    float* rois  = dets + 1152;

    // -------- workspace bump allocator --------
    char* wsp = (char*)d_ws;
    auto alloc = [&](size_t bytes) -> void* {
        void* p = (void*)wsp;
        wsp += (bytes + 255) & ~(size_t)255;
        return p;
    };
    unsigned short* Wbf1 = (unsigned short*)alloc((size_t)256 * KTOT * 2);
    unsigned short* Wbf2 = (unsigned short*)alloc((size_t)256 * KTOT * 2);
    unsigned short* Wrpn = (unsigned short*)alloc((size_t)16  * KTOT * 2);
    unsigned short* zbuf = (unsigned short*)alloc(256);
    float* inv1  = (float*)alloc(256 * 4); float* beta1 = (float*)alloc(256 * 4);
    float* inv2  = (float*)alloc(256 * 4); float* beta2 = (float*)alloc(256 * 4);
    float* l2    = (float*)alloc((size_t)2 * 256 * 4096 * 4);
    float* l3    = (float*)alloc((size_t)2 * 256 * 1024 * 4);
    unsigned short* l2bf   = (unsigned short*)alloc((size_t)2 * 256 * 4096 * 2);
    unsigned short* l3bf   = (unsigned short*)alloc((size_t)2 * 256 * 1024 * 2);
    unsigned short* l4bf   = (unsigned short*)alloc((size_t)2 * 256 * 256  * 2);
    unsigned short* tmpbf  = (unsigned short*)alloc((size_t)2 * 256 * 4096 * 2); // reused
    unsigned short* featbf = (unsigned short*)alloc((size_t)2 * 256 * 4096 * 2); // reused
    float* feat2 = (float*)alloc((size_t)2 * 256 * 4096 * 4);
    float* feat3 = (float*)alloc((size_t)2 * 256 * 1024 * 4);
    float* feat4 = (float*)alloc((size_t)2 * 256 * 256  * 4);
    float* rpnb  = (float*)alloc((size_t)2 * 16 * 4096 * 4);    // reused per level
    float* scoreB2 = (float*)alloc((size_t)2 * 12288 * 4);
    float* scoreB3 = (float*)alloc((size_t)2 * 3072  * 4);
    float* scoreB4 = (float*)alloc((size_t)2 * 768   * 4);
    float* boxB2 = (float*)alloc((size_t)2 * 12288 * 16);
    float* boxB3 = (float*)alloc((size_t)2 * 3072  * 16);
    float* boxB4 = (float*)alloc((size_t)2 * 768   * 16);
    int* coords  = (int*)alloc((size_t)2 * 3 * 32 * 4 * 4);
    int* mflags  = (int*)alloc((size_t)2 * 3 * 32 * 4);

    // -------- prep (re-run every call for determinism) --------
    k_prep_w<<<(256 * KTOT + 255) / 256, 256, 0, stream>>>(hw1, Wbf1, 256, 0);
    k_prep_w<<<(256 * KTOT + 255) / 256, 256, 0, stream>>>(hw2, Wbf2, 256, 0);
    k_prep_w<<<(3  * KTOT + 255) / 256, 256, 0, stream>>>(clsw, Wrpn, 3, 0);
    k_prep_w<<<(12 * KTOT + 255) / 256, 256, 0, stream>>>(locw, Wrpn, 12, 3);
    k_zero_u16<<<(KTOT + 255) / 256, 256, 0, stream>>>(Wrpn + (size_t)15 * KTOT, KTOT);
    k_zero_u16<<<1, 128, 0, stream>>>(zbuf, 128);
    k_prep_bn<<<1, 256, 0, stream>>>(hg1, hb1, hm1, hv1, inv1, beta1);
    k_prep_bn<<<1, 256, 0, stream>>>(hg2, hb2, hm2, hv2, inv2, beta2);

    // -------- FPN merge + one-shot bf16 channel-last conversion --------
    {
        size_t t3 = (size_t)2 * 256 * 1024;
        k_upadd<<<(unsigned)((t3 + 255) / 256), 256, 0, stream>>>(layer3, layer4, l3, 512, 32, 32);
        size_t t2 = (size_t)2 * 256 * 4096;
        k_upadd<<<(unsigned)((t2 + 255) / 256), 256, 0, stream>>>(layer2, l3, l2, 512, 64, 64);
        size_t t4 = (size_t)2 * 256 * 256;
        k_tobf16t<<<(unsigned)((t2 + 255) / 256), 256, 0, stream>>>(l2, l2bf, 4096);
        k_tobf16t<<<(unsigned)((t3 + 255) / 256), 256, 0, stream>>>(l3, l3bf, 1024);
        k_tobf16t<<<(unsigned)((t4 + 255) / 256), 256, 0, stream>>>(layer4, l4bf, 256);
    }

    // -------- per level: conv head (WMMA) + RPN + postproc --------
    const unsigned short* inbf[3] = {l2bf, l3bf, l4bf};
    float* feats[3]      = {feat2, feat3, feat4};
    float* scoreBs[3]    = {scoreB2, scoreB3, scoreB4};
    float* boxBs[3]      = {boxB2, boxB3, boxB4};
    const int  Hs[3]     = {64, 32, 16};
    const int  lgWs[3]   = {6, 5, 4};
    const int  shifts[3] = {3, 4, 5};
    const float strides[3] = {8.f, 16.f, 32.f};

    for (int lvl = 0; lvl < 3; ++lvl) {
        int H = Hs[lvl], W = H, HW = H * W, lgW = lgWs[lvl];
        dim3 g256(HW / 32, 4, 2);
        // conv1: bf16 in -> bf16 channel-last out only (tmp)
        conv3x3_wmma<64><<<g256, 256, 0, stream>>>(inbf[lvl], Wbf1, zbuf, inv1, beta1,
                                                   nullptr, tmpbf,
                                                   H, W, lgW, HW, 256, 1);
        // conv2: bf16 in -> f32 NCHW feat (ROI pool) + bf16 chan-last (RPN conv)
        conv3x3_wmma<64><<<g256, 256, 0, stream>>>(tmpbf, Wbf2, zbuf, inv2, beta2,
                                                   feats[lvl], featbf,
                                                   H, W, lgW, HW, 256, 1);
        // rpn conv: 16 output rows (3 cls + 12 loc + pad), raw f32 out
        dim3 g16(HW / 32, 1, 2);
        conv3x3_wmma<16><<<g16, 64, 0, stream>>>(featbf, Wrpn, zbuf, nullptr, nullptr,
                                                 rpnb, nullptr,
                                                 H, W, lgW, HW, 16, 0);
        int A = HW * 3;
        dim3 gd((A + 255) / 256, 2);
        k_decode<<<gd, 256, 0, stream>>>(rpnb, anch[lvl], scoreBs[lvl], boxBs[lvl],
                                         HW, strides[lvl]);
        k_topk_nms<<<2, 256, 0, stream>>>(scoreBs[lvl], boxBs[lvl], A, lvl, shifts[lvl],
                                          dets, maskO, coords, mflags);
        k_roipool<<<dim3(32, 2), 256, 0, stream>>>(feats[lvl], coords, mflags,
                                                   rois, H, W, lvl);
    }
}
